// LinearChainCRF_54434415509671
// MI455X (gfx1250) — matrix operand order repaired and checked by reading, not compiled, not run on hardware
//
#include <hip/hip_runtime.h>
#include <hip/hip_bf16.h>

typedef __attribute__((ext_vector_type(16))) _Float16 v16h;
typedef __attribute__((ext_vector_type(8)))  float    v8f;
typedef __attribute__((ext_vector_type(4)))  int      v4i;

#define CRF_B 128
#define CRF_T 4096
#define CRF_N 48

// 4095 scan steps = 585 chunks x 7 steps (exact)
#define CRF_CHUNK  7
#define CRF_NCHUNK 585

union FragH { v16h h; float4 q[2]; };

// ---------------- CDNA5 async global->LDS path (guarded) --------------------
#if defined(__HIP_DEVICE_COMPILE__) && __has_builtin(__builtin_amdgcn_global_load_async_to_lds_b128)
#define CRF_ASYNC 1
#else
#define CRF_ASYNC 0
#endif

#if CRF_ASYNC
typedef __attribute__((address_space(1))) v4i* gv4i_p;   // global v4i*
typedef __attribute__((address_space(3))) v4i* lv4i_p;   // LDS v4i*
__device__ __forceinline__ void crf_async_b128(const float* src, float* lds_dst) {
    // low 32 bits of a generic LDS pointer are the LDS byte offset (ISA 10.2)
    __builtin_amdgcn_global_load_async_to_lds_b128(
        (gv4i_p)(unsigned long long)src,
        (lv4i_p)(unsigned int)(unsigned long long)lds_dst,
        0, 0);
}
#if __has_builtin(__builtin_amdgcn_s_wait_asynccnt)
#define CRF_WAIT_ASYNC(n) __builtin_amdgcn_s_wait_asynccnt(n)
#else
#define CRF_WAIT_ASYNC(n) asm volatile("s_wait_asynccnt %0" ::"i"(n) : "memory")
#endif
#else
#define CRF_WAIT_ASYNC(n)
#endif

// fast log2 (raw v_log_f32); inputs are in [~0.9, 48] so no denorm guard needed
__device__ __forceinline__ float crf_log2(float x) {
#if defined(__HIP_DEVICE_COMPILE__) && __has_builtin(__builtin_amdgcn_logf)
    return __builtin_amdgcn_logf(x);
#else
    return __log2f(x);
#endif
}
#define CRF_LN2 0.6931471805599453f

// ---------------------------------------------------------------------------
// Kernel 1: forward partition (log Z) via WMMA.
// Grid = 8 blocks (16-batch tiles), block = 96 threads = 3 waves (wave32).
// Wave w owns output tag tile [w*16, w*16+16).
// Per step:  m[b] = max_i alpha[b,i];  aExp = exp(alpha-m) in f16 (K pad 48->64)
//            S = aExp @ exp(trans)  (2x v_wmma_f32_16x16x32_f16)
//            alpha'[b,j] = m[b] + ln2*log2(S) + em[b,t,j]
// Emissions stream in via double-buffered async-to-LDS chunks of 7 steps.
// ---------------------------------------------------------------------------
__global__ __launch_bounds__(96)
void crf_forward_wmma(const float* __restrict__ emissions,
                      const float* __restrict__ transitions,
                      const float* __restrict__ startv,
                      const float* __restrict__ endv,
                      float* __restrict__ logZ)
{
    __shared__ __align__(16) float    sEmBuf[2][CRF_CHUNK][16][48]; // 42 KB staging
    __shared__ __align__(16) float    sAlpha[16][48];
    __shared__ __align__(16) _Float16 sAExp[16][64];
    __shared__ __align__(16) _Float16 sTransT[48][64];
    __shared__ float sM[16];

    const int tid   = threadIdx.x;
    const int lane  = tid & 31;
    const int wave  = tid >> 5;                 // 0..2 : tag tile
    const int bBase = blockIdx.x * 16;
    const long long TN = (long long)CRF_T * CRF_N;

    // ---- one-time LDS init ----
    for (int idx = tid; idx < 48 * 64; idx += 96) {
        int col = idx >> 6, k = idx & 63;
        sTransT[col][k] = (k < CRF_N) ? (_Float16)__expf(transitions[k * CRF_N + col])
                                      : (_Float16)0.0f;
    }
    for (int idx = tid; idx < 16 * 64; idx += 96)
        sAExp[idx >> 6][idx & 63] = (_Float16)0.0f;
    for (int idx = tid; idx < 16 * 48; idx += 96) {
        int b = idx / 48, j = idx % 48;
        sAlpha[b][j] = startv[j] + emissions[(long long)(bBase + b) * TN + j];
    }
    __syncthreads();

    const int n  = lane & 15;
    const int hi = lane >> 4;

    // ---- constant B fragments (exp-transition tile for this wave) ----
    FragH fb0, fb1;
    {
        const _Float16* bp = &sTransT[wave * 16 + n][hi * 16];
        fb0.q[0] = *(const float4*)(bp);
        fb0.q[1] = *(const float4*)(bp + 8);
        fb1.q[0] = *(const float4*)(bp + 32);
        fb1.q[1] = *(const float4*)(bp + 40);
    }

    // per-thread staging slice: batch eb, 8 floats from ek (two B128 per step)
    const int eb = tid / 6;
    const int ek = (tid % 6) * 8;
    const float* emPtr = emissions + (long long)(bBase + eb) * TN + ek;

    const int pb = tid & 15;                    // exp/max mapping: batch
    const int pg = tid >> 4;                    // tag group (8 tags)

#if CRF_ASYNC
    // prologue: chunks 0 and 1 in flight (14 async B128 per thread per chunk)
    #pragma unroll
    for (int s = 0; s < CRF_CHUNK; ++s) {
        const float* src = emPtr + (long long)(1 + s) * CRF_N;
        crf_async_b128(src,     &sEmBuf[0][s][eb][ek]);
        crf_async_b128(src + 4, &sEmBuf[0][s][eb][ek + 4]);
    }
    #pragma unroll
    for (int s = 0; s < CRF_CHUNK; ++s) {
        const float* src = emPtr + (long long)(1 + CRF_CHUNK + s) * CRF_N;
        crf_async_b128(src,     &sEmBuf[1][s][eb][ek]);
        crf_async_b128(src + 4, &sEmBuf[1][s][eb][ek + 4]);
    }
#endif

    for (int c = 0; c < CRF_NCHUNK; ++c) {
        const int buf = c & 1;
#if CRF_ASYNC
        if (c + 1 < CRF_NCHUNK) { CRF_WAIT_ASYNC(14); }   // chunk c landed
        else                    { CRF_WAIT_ASYNC(0);  }
#else
        // synchronous fallback staging
        for (int s = 0; s < CRF_CHUNK; ++s) {
            const float* src = emPtr + (long long)(1 + c * CRF_CHUNK + s) * CRF_N;
            *(float4*)&sEmBuf[buf][s][eb][ek]     = *(const float4*)(src);
            *(float4*)&sEmBuf[buf][s][eb][ek + 4] = *(const float4*)(src + 4);
        }
#endif
        __syncthreads();

        for (int s = 0; s < CRF_CHUNK; ++s) {
            // ---- phase A: per-batch max (vectorized, redundant x6) + exp ----
            {
                float4 v0 = *(const float4*)&sAlpha[pb][0];
                float  m  = fmaxf(fmaxf(v0.x, v0.y), fmaxf(v0.z, v0.w));
                #pragma unroll
                for (int q = 1; q < 12; ++q) {
                    float4 v = *(const float4*)&sAlpha[pb][q * 4];
                    m = fmaxf(m, fmaxf(fmaxf(v.x, v.y), fmaxf(v.z, v.w)));
                }
                if (pg == 0) sM[pb] = m;
                float4 e0 = *(const float4*)&sAlpha[pb][pg * 8];
                float4 e1 = *(const float4*)&sAlpha[pb][pg * 8 + 4];
                _Float16* dst = &sAExp[pb][pg * 8];
                dst[0] = (_Float16)__expf(e0.x - m);
                dst[1] = (_Float16)__expf(e0.y - m);
                dst[2] = (_Float16)__expf(e0.z - m);
                dst[3] = (_Float16)__expf(e0.w - m);
                dst[4] = (_Float16)__expf(e1.x - m);
                dst[5] = (_Float16)__expf(e1.y - m);
                dst[6] = (_Float16)__expf(e1.z - m);
                dst[7] = (_Float16)__expf(e1.w - m);
            }
            __syncthreads();

            // ---- phase B: WMMA GEMM + fast log + emission; update alpha ----
            {
                FragH fa0, fa1;
                const _Float16* ap = &sAExp[n][hi * 16];
                fa0.q[0] = *(const float4*)(ap);
                fa0.q[1] = *(const float4*)(ap + 8);
                fa1.q[0] = *(const float4*)(ap + 32);
                fa1.q[1] = *(const float4*)(ap + 40);

                v8f acc = {};
                acc = __builtin_amdgcn_wmma_f32_16x16x32_f16(
                          false, fa0.h, false, fb0.h, (short)0, acc, false, false);
                acc = __builtin_amdgcn_wmma_f32_16x16x32_f16(
                          false, fa1.h, false, fb1.h, (short)0, acc, false, false);

                const int j = wave * 16 + n;
                #pragma unroll
                for (int v = 0; v < 8; ++v) {
                    int M = hi * 8 + v;
                    sAlpha[M][j] = sM[M] + CRF_LN2 * crf_log2(acc[v])
                                 + sEmBuf[buf][s][M][j];
                }
            }
            __syncthreads();
        }

#if CRF_ASYNC
        if (c + 2 < CRF_NCHUNK) {       // refill the buffer we just drained
            #pragma unroll
            for (int s = 0; s < CRF_CHUNK; ++s) {
                const float* src = emPtr + (long long)(1 + (c + 2) * CRF_CHUNK + s) * CRF_N;
                crf_async_b128(src,     &sEmBuf[buf][s][eb][ek]);
                crf_async_b128(src + 4, &sEmBuf[buf][s][eb][ek + 4]);
            }
        }
#endif
    }

    // ---- finalize: logZ[b] = logsumexp_j(alpha[b,j] + end[j]) ----
    if (tid < 16) {
        int b = tid;
        float m = sAlpha[b][0] + endv[0];
        for (int j = 1; j < CRF_N; ++j) m = fmaxf(m, sAlpha[b][j] + endv[j]);
        float ssum = 0.0f;
        for (int j = 0; j < CRF_N; ++j) ssum += __expf(sAlpha[b][j] + endv[j] - m);
        logZ[bBase + b] = m + CRF_LN2 * crf_log2(ssum);
    }
}

// ---------------------------------------------------------------------------
// Kernel 2: gold-path score + final NLL. One block per sequence.
// ---------------------------------------------------------------------------
__global__ __launch_bounds__(256)
void crf_gold_score(const float* __restrict__ emissions,
                    const float* __restrict__ transitions,
                    const float* __restrict__ startv,
                    const float* __restrict__ endv,
                    const int*   __restrict__ tags,
                    const float* __restrict__ logZ,
                    float* __restrict__ out)
{
    const int b = blockIdx.x;
    const long long TN = (long long)CRF_T * CRF_N;
    const float* emB = emissions + (long long)b * TN;
    const int*   tg  = tags + (long long)b * CRF_T;

    float acc = 0.0f;
    for (int t = threadIdx.x; t < CRF_T; t += 256) {
        int tag = tg[t];
        acc += emB[(long long)t * CRF_N + tag];
        if (t + 1 < CRF_T) acc += transitions[tag * CRF_N + tg[t + 1]];
    }

    __shared__ float red[256];
    red[threadIdx.x] = acc;
    __syncthreads();
    #pragma unroll
    for (int s = 128; s > 0; s >>= 1) {
        if (threadIdx.x < s) red[threadIdx.x] += red[threadIdx.x + s];
        __syncthreads();
    }
    if (threadIdx.x == 0) {
        float score = red[0] + startv[tg[0]] + endv[tg[CRF_T - 1]];
        out[b] = logZ[b] - score;
    }
}

// ---------------------------------------------------------------------------
extern "C" void kernel_launch(void* const* d_in, const int* in_sizes, int n_in,
                              void* d_out, int out_size, void* d_ws, size_t ws_size,
                              hipStream_t stream) {
    const float* emissions   = (const float*)d_in[0];
    const float* transitions = (const float*)d_in[1];
    const float* startv      = (const float*)d_in[2];
    const float* endv        = (const float*)d_in[3];
    const int*   tags        = (const int*)d_in[4];
    float* out  = (float*)d_out;
    float* logZ = (float*)d_ws;     // 128 floats of scratch

    crf_forward_wmma<<<CRF_B / 16, 96, 0, stream>>>(emissions, transitions,
                                                    startv, endv, logZ);
    crf_gold_score<<<CRF_B, 256, 0, stream>>>(emissions, transitions, startv,
                                              endv, tags, logZ, out);
}
